// MultiHeadAttentionBlock_60189671686492
// MI455X (gfx1250) — matrix operand order repaired and checked by reading, not compiled, not run on hardware
//
#include <hip/hip_runtime.h>

// ---------------------------------------------------------------------------
// MultiHeadAttention forward for MI455X (gfx1250, wave32, WMMA + TDM).
// All matmuls demoted to bf16 with f32 accumulation (v_wmma_f32_16x16x32_bf16).
// Weight/V tiles staged to LDS with the Tensor Data Mover (tensor_load_to_lds).
// Softmax row-sums computed on the matrix pipe via an all-ones WMMA column.
// ---------------------------------------------------------------------------

typedef __attribute__((ext_vector_type(16))) __bf16 v16bf;
typedef __attribute__((ext_vector_type(8)))  float  v8f;
typedef __attribute__((ext_vector_type(4)))  unsigned int u32x4;
typedef __attribute__((ext_vector_type(8)))  int  i32x8;
typedef __attribute__((ext_vector_type(4)))  int  i32x4;

#if __has_builtin(__builtin_amdgcn_tensor_load_to_lds) && __has_builtin(__builtin_amdgcn_s_wait_tensorcnt)
#define HAVE_TDM 1
#else
#define HAVE_TDM 0
#endif

union Frag  { uint4 u[2]; v16bf v; };
union FragH { unsigned short h[16]; v16bf v; };

__device__ __forceinline__ unsigned short f2bf(float f) {
  unsigned int u = __float_as_uint(f);
  u = u + 0x7FFFu + ((u >> 16) & 1u);   // round-to-nearest-even
  return (unsigned short)(u >> 16);
}

__device__ __forceinline__ v8f zero8() {
  v8f z;
#pragma unroll
  for (int i = 0; i < 8; ++i) z[i] = 0.0f;
  return z;
}

// A-fragment (16x32 bf16, MxK): lane<16 -> row M=lane, K = 16h .. 16h+15 contiguous
__device__ __forceinline__ v16bf load_a_frag(const unsigned short* A, int ld, int lane) {
  const int half = lane >> 4, row = lane & 15;
  const unsigned short* p = A + row * ld + 16 * half;
  Frag f;
  f.u[0] = *reinterpret_cast<const uint4*>(p);        // K = 16h .. 16h+7
  f.u[1] = *reinterpret_cast<const uint4*>(p + 8);    // K = 16h+8 .. 16h+15
  return f.v;
}

// B-fragment (32x16 bf16, KxN) sourced from a row-major [N][K] matrix Bm
// (B[k][n] = Bm[n*ld + k]): lane<16 -> col N=lane, K = 16h .. 16h+15 contiguous.
__device__ __forceinline__ v16bf load_b_frag(const unsigned short* Bm, int ld, int lane) {
  const int half = lane >> 4, col = lane & 15;
  const unsigned short* p = Bm + col * ld + 16 * half;
  Frag f;
  f.u[0] = reinterpret_cast<const uint4*>(p)[0];
  f.u[1] = reinterpret_cast<const uint4*>(p)[1];
  return f.v;
}

// B-fragment gathered from a row-major [K][N] tile (B[k][n] = Bm[k*ld + n]).
__device__ __forceinline__ v16bf load_b_frag_kn(const unsigned short* Bm, int ld, int lane) {
  const int half = lane >> 4, col = lane & 15;
  FragH f;
#pragma unroll
  for (int kk = 0; kk < 16; ++kk)
    f.h[kk] = Bm[(16 * half + kk) * ld + col];
  return f.v;
}

// All-ones bf16 B fragment (for row-sum-via-WMMA).
__device__ __forceinline__ v16bf ones_frag() {
  FragH f;
#pragma unroll
  for (int i = 0; i < 16; ++i) f.h[i] = 0x3F80u;      // bf16 1.0
  return f.v;
}

__device__ __forceinline__ v8f wmma_bf16(v16bf a, v16bf b, v8f c) {
  return __builtin_amdgcn_wmma_f32_16x16x32_bf16(
      /*neg_a=*/false, a, /*neg_b=*/false, b,
      /*c_mod=*/(short)0, c, /*reuse_a=*/false, /*reuse_b=*/false);
}

// ---------------------------------------------------------------------------
// TDM: DMA a 2-D bf16 tile (td0 cols x td1 rows, row stride st0 elements)
// from global memory into LDS at byte offset lds_off.  D# per CDNA5 ISA §8.
// ---------------------------------------------------------------------------
__device__ __forceinline__ void tdm_load_2d(unsigned lds_off, const void* gsrc,
                                            unsigned td0, unsigned td1,
                                            unsigned long long st0) {
#if HAVE_TDM
  const unsigned long long ga = (unsigned long long)(size_t)gsrc;
  u32x4 g0;
  g0[0] = 1u;                                                  // count=1, user D#
  g0[1] = lds_off;                                             // lds_addr
  g0[2] = (unsigned)ga;                                        // global_addr[31:0]
  g0[3] = (unsigned)((ga >> 32) & 0x01FFFFFFull) | 0x80000000u;// addr[56:32] | type=2
  // group1 bit assembly (bits 0..255 as four qwords)
  const unsigned long long q0 = (1ull << 16)                               // data_size=2B
                              | ((unsigned long long)(td0 & 0xFFFFu) << 48);// tensor_dim0[15:0]
  const unsigned long long q1 = ((unsigned long long)(td0 >> 16))           // tensor_dim0[31:16]
                              | ((unsigned long long)td1 << 16)             // tensor_dim1
                              | ((unsigned long long)(td0 & 0xFFFFu) << 48);// tile_dim0
  const unsigned long long q2 = ((unsigned long long)(td1 & 0xFFFFu))       // tile_dim1
                              | ((st0 & 0xFFFFFFFFull) << 32);              // dim0_stride[31:0]
  const unsigned long long q3 = ((st0 >> 32) & 0xFFFFull);                  // dim0_stride[47:32]
  i32x8 g1;
  g1[0] = (int)q0; g1[1] = (int)(q0 >> 32);
  g1[2] = (int)q1; g1[3] = (int)(q1 >> 32);
  g1[4] = (int)q2; g1[5] = (int)(q2 >> 32);
  g1[6] = (int)q3; g1[7] = (int)(q3 >> 32);
  i32x4 gz4; gz4[0] = 0; gz4[1] = 0; gz4[2] = 0; gz4[3] = 0;
  i32x8 gz8;
#pragma unroll
  for (int t = 0; t < 8; ++t) gz8[t] = 0;
  __builtin_amdgcn_tensor_load_to_lds(g0, g1, gz4, gz4, gz8, 0);
#endif
}

// ---------------------------------------------------------------------------
// Kernel 1: fp32 -> bf16 conversion (4 elements / thread)
// ---------------------------------------------------------------------------
__global__ void cvt_f32_bf16(const float* __restrict__ in,
                             unsigned short* __restrict__ out, int n4) {
  int i = blockIdx.x * blockDim.x + threadIdx.x;
  if (i >= n4) return;
  float4 f = reinterpret_cast<const float4*>(in)[i];
  ushort4 o;
  o.x = f2bf(f.x); o.y = f2bf(f.y); o.z = f2bf(f.z); o.w = f2bf(f.w);
  reinterpret_cast<ushort4*>(out)[i] = o;
}

// ---------------------------------------------------------------------------
// Kernel 2: Y = A(MxK bf16) @ W^T (W is NxK bf16) + bias
//   EPI==0 : write bf16, reshaped to [B, H, S, d_k] (projection path)
//   EPI==1 : write fp32, row-major [M, N]           (output projection)
// Block = 128 threads (4 waves); wave tile 32(M) x 64(N); block tile 128 x 64.
// The shared 64x32 W tile is TDM-DMA'd into LDS (double-buffered) by wave 0;
// all 4 waves consume it, so each W tile is fetched from global exactly once.
// ---------------------------------------------------------------------------
template <int EPI>
__global__ void gemm_bias(const unsigned short* __restrict__ A,
                          const unsigned short* __restrict__ W,
                          const float* __restrict__ bias,
                          void* __restrict__ out,
                          int M, int N, int K, int S) {
  constexpr int H = 16;
  const int lane = threadIdx.x & 31;
  const int wave = threadIdx.x >> 5;            // 0..3
  const int m0 = blockIdx.y * 128 + wave * 32;
  const int n0 = blockIdx.x * 64;

  __shared__ __align__(16) unsigned short Wt[2][64 * 32];   // [buf][n][k], ld=32
  const unsigned wt_lds = (unsigned)(size_t)(void*)&Wt[0][0];

  v8f acc[2][4];
#pragma unroll
  for (int rg = 0; rg < 2; ++rg)
#pragma unroll
    for (int nf = 0; nf < 4; ++nf) acc[rg][nf] = zero8();

  const unsigned short* A0 = A + (size_t)m0 * K;
  const unsigned short* A1 = A0 + (size_t)16 * K;
  const unsigned short* Wb = W + (size_t)n0 * K;

#if HAVE_TDM
  if (wave == 0)
    tdm_load_2d(wt_lds, Wb, /*td0=*/32, /*td1=*/64, /*st0=*/K);
  __builtin_amdgcn_s_wait_tensorcnt(0);
  __syncthreads();
#endif

  for (int k0 = 0; k0 < K; k0 += 32) {
#if HAVE_TDM
    const int cur = (k0 >> 5) & 1;
    if (wave == 0 && k0 + 32 < K)   // prefetch next tile into the other buffer
      tdm_load_2d(wt_lds + (unsigned)((cur ^ 1) * 64 * 32 * 2), Wb + k0 + 32,
                  /*td0=*/32, /*td1=*/64, /*st0=*/K);
    const unsigned short* Wk = &Wt[cur][0];
#endif
    __builtin_prefetch(A0 + k0 + 128, 0, 1);
    v16bf a0 = load_a_frag(A0 + k0, K, lane);
    v16bf a1 = load_a_frag(A1 + k0, K, lane);
#pragma unroll
    for (int nf = 0; nf < 4; ++nf) {
#if HAVE_TDM
      v16bf b = load_b_frag(Wk + nf * 16 * 32, 32, lane);
#else
      v16bf b = load_b_frag(Wb + (size_t)(nf * 16) * K + k0, K, lane);
#endif
      acc[0][nf] = wmma_bf16(a0, b, acc[0][nf]);
      acc[1][nf] = wmma_bf16(a1, b, acc[1][nf]);
    }
#if HAVE_TDM
    __builtin_amdgcn_s_wait_tensorcnt(0);   // next tile resident
    __syncthreads();
#endif
  }

  // C/D layout: vgpr i, lane j -> (M = i + 8*(j>>4), N = j&15)
  const int c = lane & 15;
#pragma unroll
  for (int rg = 0; rg < 2; ++rg) {
    const int rbase = m0 + rg * 16 + ((lane >> 4) << 3);
#pragma unroll
    for (int nf = 0; nf < 4; ++nf) {
      const int n  = n0 + nf * 16 + c;
      const float bv = bias[n];
#pragma unroll
      for (int i = 0; i < 8; ++i) {
        const int m = rbase + i;
        const float val = acc[rg][nf][i] + bv;
        if (EPI == 0) {
          const int b = m / S, s = m - b * S;
          const int h = n >> 6, d = n & 63;
          ((unsigned short*)out)[(((size_t)(b * H + h)) * S + s) * 64 + d] = f2bf(val);
        } else {
          ((float*)out)[(size_t)m * N + n] = val;
        }
      }
    }
  }
}

// ---------------------------------------------------------------------------
// Kernel 3: flash attention, bf16 WMMA, online softmax, TDM-staged V tiles.
// grid = (S/16, B*H), block = 32 (one wave per 16 query rows).
// Softmax denominator is accumulated as a 5th WMMA accumulator (P @ ones),
// so only the row-max needs a cross-lane shuffle reduction.
// ---------------------------------------------------------------------------
__global__ void flash_attn(const unsigned short* __restrict__ Q,
                           const unsigned short* __restrict__ K,
                           const unsigned short* __restrict__ V,
                           unsigned short* __restrict__ O,
                           int S) {
  constexpr int DK = 64, H = 16, D = 1024;
  const int lane = threadIdx.x & 31;
  const int qb = blockIdx.x;
  const int bh = blockIdx.y;
  const int b = bh / H, h = bh - b * H;

  const unsigned short* Qp = Q + ((size_t)bh * S + qb * 16) * DK;
  const unsigned short* Kp = K + (size_t)bh * S * DK;
  const unsigned short* Vp = V + (size_t)bh * S * DK;

  __shared__ __align__(16) unsigned short Vtile[32 * 64];  // row-major [k][d] via TDM
  __shared__ __align__(16) unsigned short Plds[16 * 32];   // P tile, row-major 16x32

  const v16bf aq0 = load_a_frag(Qp, DK, lane);             // d 0..31
  const v16bf aq1 = load_a_frag(Qp + 32, DK, lane);        // d 32..63
  const v16bf ones = ones_frag();

  v8f o[5];                                                // [0..3]=O cols, [4]=row sums
#pragma unroll
  for (int nf = 0; nf < 5; ++nf) o[nf] = zero8();
  float mrun[8];
#pragma unroll
  for (int i = 0; i < 8; ++i) mrun[i] = -3.0e38f;

  const float scale = 0.125f;                              // 1/sqrt(64)
  const int c = lane & 15, half = lane >> 4;
  const unsigned vt_lds = (unsigned)(size_t)(void*)&Vtile[0];

  for (int j0 = 0; j0 < S; j0 += 32) {
    // ---- kick off async DMA of the V tile; overlaps with QK^T + softmax ----
#if HAVE_TDM
    tdm_load_2d(vt_lds, Vp + (size_t)j0 * DK, /*td0=*/DK, /*td1=*/32, /*st0=*/DK);
#else
    {
      const uint4* src = reinterpret_cast<const uint4*>(Vp + (size_t)(j0 + lane) * DK);
      uint4* dst = reinterpret_cast<uint4*>(&Vtile[lane * DK]);
#pragma unroll
      for (int t = 0; t < 8; ++t) dst[t] = src[t];
    }
#endif

    // ---- scores: two 16x16 tiles (keys j0..j0+15 and j0+16..j0+31) ----
    v8f s0 = zero8(), s1 = zero8();
    s0 = wmma_bf16(aq0, load_b_frag(Kp + (size_t)j0 * DK, DK, lane), s0);
    s0 = wmma_bf16(aq1, load_b_frag(Kp + (size_t)j0 * DK + 32, DK, lane), s0);
    s1 = wmma_bf16(aq0, load_b_frag(Kp + (size_t)(j0 + 16) * DK, DK, lane), s1);
    s1 = wmma_bf16(aq1, load_b_frag(Kp + (size_t)(j0 + 16) * DK + 32, DK, lane), s1);

    // ---- online softmax update: row-max shuffle, exp, alpha-rescale ----
#pragma unroll
    for (int i = 0; i < 8; ++i) {
      float v0 = s0[i] * scale, v1 = s1[i] * scale;
      float mx = fmaxf(v0, v1);
#pragma unroll
      for (int off = 1; off < 16; off <<= 1) mx = fmaxf(mx, __shfl_xor(mx, off, 32));
      const float mnew = fmaxf(mrun[i], mx);
      const float p0 = __expf(v0 - mnew);
      const float p1 = __expf(v1 - mnew);
      const float alpha = __expf(mrun[i] - mnew);
      mrun[i] = mnew;
#pragma unroll
      for (int nf = 0; nf < 5; ++nf) o[nf][i] *= alpha;    // rescale O and row-sum acc
      const int r = (half << 3) + i;
      Plds[r * 32 + c]      = f2bf(p0);
      Plds[r * 32 + 16 + c] = f2bf(p1);
    }

#if HAVE_TDM
    __builtin_amdgcn_s_wait_tensorcnt(0);   // V tile resident in LDS
#endif
    __syncthreads();

    // ---- O += P(16x32) @ V(32x64);  rowsum += P @ ones ----
    const v16bf ap = load_a_frag(Plds, 32, lane);
#pragma unroll
    for (int nf = 0; nf < 4; ++nf) {
      v16bf bv = load_b_frag_kn(Vtile + nf * 16, DK, lane);
      o[nf] = wmma_bf16(ap, bv, o[nf]);
    }
    o[4] = wmma_bf16(ap, ones, o[4]);       // softmax denominator on the matrix pipe
    __syncthreads();
  }

  // ---- normalize and write, merging heads back to [B, S, D] ----
#pragma unroll
  for (int nf = 0; nf < 4; ++nf) {
#pragma unroll
    for (int i = 0; i < 8; ++i) {
      const int r = (half << 3) + i;
      const int s = qb * 16 + r;
      const int d = nf * 16 + c;
      O[((size_t)b * S + s) * D + h * DK + d] = f2bf(o[nf][i] / o[4][i]);
    }
  }
}

// ---------------------------------------------------------------------------
// Host-side orchestration
// ---------------------------------------------------------------------------
extern "C" void kernel_launch(void* const* d_in, const int* in_sizes, int n_in,
                              void* d_out, int out_size, void* d_ws, size_t ws_size,
                              hipStream_t stream) {
  const float* q   = (const float*)d_in[0];
  const float* k   = (const float*)d_in[1];
  const float* v   = (const float*)d_in[2];
  const float* w_q = (const float*)d_in[3];
  const float* b_q = (const float*)d_in[4];
  const float* w_k = (const float*)d_in[5];
  const float* b_k = (const float*)d_in[6];
  const float* w_v = (const float*)d_in[7];
  const float* b_v = (const float*)d_in[8];
  const float* w_o = (const float*)d_in[9];
  const float* b_o = (const float*)d_in[10];

  const int D = 1024;
  const int M = in_sizes[0] / D;    // B*S = 4096
  const int Bb = 2;
  const int S = M / Bb;             // 2048

  // Workspace layout (all bf16 as ushort): ~64 MB total
  unsigned short* ws = (unsigned short*)d_ws;
  size_t off = 0;
  auto take = [&](size_t n) { unsigned short* p = ws + off; off += n; return p; };
  unsigned short* qbf  = take((size_t)M * D);
  unsigned short* kbf  = take((size_t)M * D);
  unsigned short* vbf  = take((size_t)M * D);
  unsigned short* wqb  = take((size_t)D * D);
  unsigned short* wkb  = take((size_t)D * D);
  unsigned short* wvb  = take((size_t)D * D);
  unsigned short* wob  = take((size_t)D * D);
  unsigned short* Qp   = take((size_t)M * D);   // [B,H,S,64]
  unsigned short* Kp   = take((size_t)M * D);
  unsigned short* Vp   = take((size_t)M * D);
  unsigned short* attn = take((size_t)M * D);   // [B,S,D]

  // 1) convert to bf16
  {
    const int n4a = M * D / 4;                  // activations
    const int n4w = D * D / 4;                  // weights
    dim3 blk(256);
    cvt_f32_bf16<<<(n4a + 255) / 256, blk, 0, stream>>>(q, qbf, n4a);
    cvt_f32_bf16<<<(n4a + 255) / 256, blk, 0, stream>>>(k, kbf, n4a);
    cvt_f32_bf16<<<(n4a + 255) / 256, blk, 0, stream>>>(v, vbf, n4a);
    cvt_f32_bf16<<<(n4w + 255) / 256, blk, 0, stream>>>(w_q, wqb, n4w);
    cvt_f32_bf16<<<(n4w + 255) / 256, blk, 0, stream>>>(w_k, wkb, n4w);
    cvt_f32_bf16<<<(n4w + 255) / 256, blk, 0, stream>>>(w_v, wvb, n4w);
    cvt_f32_bf16<<<(n4w + 255) / 256, blk, 0, stream>>>(w_o, wob, n4w);
  }

  // 2) Q/K/V projections (bf16 WMMA, TDM-staged weights, head-split epilogue)
  {
    dim3 grid(D / 64, M / 128), blk(128);
    gemm_bias<0><<<grid, blk, 0, stream>>>(qbf, wqb, b_q, Qp, M, D, D, S);
    gemm_bias<0><<<grid, blk, 0, stream>>>(kbf, wkb, b_k, Kp, M, D, D, S);
    gemm_bias<0><<<grid, blk, 0, stream>>>(vbf, wvb, b_v, Vp, M, D, D, S);
  }

  // 3) attention (flash-style, WMMA for QK^T / PV / row-sums, TDM for V tiles)
  {
    dim3 grid(S / 16, Bb * 16), blk(32);
    flash_attn<<<grid, blk, 0, stream>>>(Qp, Kp, Vp, attn, S);
  }

  // 4) output projection -> fp32 d_out
  {
    dim3 grid(D / 64, M / 128), blk(128);
    gemm_bias<1><<<grid, blk, 0, stream>>>(attn, wob, b_o, (float*)d_out, M, D, D, S);
  }
}
